// Model_64974265254508
// MI455X (gfx1250) — compile-verified, matrix operand order checked by reading, not run on hardware
//
#include <hip/hip_runtime.h>
#include <hip/hip_bf16.h>
#include <math.h>
#include <stdint.h>

// ---------------- types ----------------
typedef __attribute__((ext_vector_type(16))) __bf16 v16bf;
typedef __attribute__((ext_vector_type(8)))  float  v8f;
typedef __attribute__((ext_vector_type(4)))  unsigned v4u;

#define D_MODEL 512
#define D_FF    2048
#define NB      8      // batch
#define LSEQ    1024   // both enc and dec token length
#define NH      8
#define DH      64
#define TOK     (NB*LSEQ)   // 8192
#define UTOP    35          // 5*ceil(ln(1024))

// GEMM tiling
#define BM 128
#define BN 64
#define BK 32
#define A_TILE_BYTES (BM * BK * 2)   // 8192
#define B_TILE_BYTES (BN * BK * 2)   // 4096
#define STAGE_BYTES  (A_TILE_BYTES + B_TILE_BYTES)

// ---------------- device helpers ----------------
__device__ __forceinline__ __bf16 tobf(float f) {
    unsigned u = __builtin_bit_cast(unsigned, f);
    u += 0x7FFFu + ((u >> 16) & 1u);           // round-to-nearest-even
    unsigned short h = (unsigned short)(u >> 16);
    return __builtin_bit_cast(__bf16, h);
}

__device__ __forceinline__ float gelu_exact(float x) {
    return 0.5f * x * (1.0f + erff(x * 0.70710678118654752f));
}

// On AMDGPU, the low 32 bits of a generic pointer to LDS are the LDS byte
// offset (ISA 10.2). The ptrtoint also forces the
// shared array's address to escape so its loads/stores cannot be folded away.
__device__ __forceinline__ unsigned lds_addr_of(const void* p) {
    return (unsigned)(size_t)p;
}

// ---------------- packing: f32 -> bf16 ----------------
// weights: f32 KxN -> bf16 NxK (transpose)
__global__ void pack_w_kernel(const float* __restrict__ W, __bf16* __restrict__ out, int K, int N) {
    int idx = blockIdx.x * blockDim.x + threadIdx.x;
    if (idx >= K * N) return;
    int n = idx / K, k = idx % K;
    out[idx] = tobf(W[(size_t)k * N + n]);
}
// activations: f32 MxK row-major -> bf16 MxK row-major
__global__ void pack_a_kernel(const float* __restrict__ A, __bf16* __restrict__ out, int total) {
    int idx = blockIdx.x * blockDim.x + threadIdx.x;
    if (idx >= total) return;
    out[idx] = tobf(A[idx]);
}

// fragment from LDS tile (row-major, 32 bf16 = 64B per row)
// element e of lane: k = (e&7) + (e>=8?16:0) + (lane>=16?8:0); row = base_row for every lane
__device__ __forceinline__ v16bf frag_lds(const char* tile, int row, int lane) {
    const char* p = tile + (unsigned)row * 64u + ((lane >> 4) << 4);
    v4u lo = *(const v4u*)(p);
    v4u hi = *(const v4u*)(p + 32);
    v16bf f;
    unsigned* fp = (unsigned*)&f;
    fp[0] = lo.x; fp[1] = lo.y; fp[2] = lo.z; fp[3] = lo.w;
    fp[4] = hi.x; fp[5] = hi.y; fp[6] = hi.z; fp[7] = hi.w;
    return f;
}

// ---------------- WMMA GEMM with async-LDS double-buffered staging ----------------
// C[M,N] = Abf[M,K](bf16,row) * Bp[N,K](bf16, N-major == W^T) + bias ; act 0=none 1=gelu
__global__ __launch_bounds__(256) void gemm_wmma_kernel(
    const __bf16* __restrict__ Abf, const __bf16* __restrict__ Bp,
    const float* __restrict__ bias, float* __restrict__ C,
    int M, int N, int K, int act)
{
    __shared__ __align__(16) char smem[2 * STAGE_BYTES];   // 24 KB, double buffered
    const int t    = threadIdx.x;
    const int lane = t & 31;
    const int bm = blockIdx.y * BM;
    const int bn = blockIdx.x * BN;
    const int wid = t >> 5;
    const int wm  = (wid >> 1) * 32;     // wave m-offset: 0,32,64,96
    const int wn  = (wid & 1) * 32;      // wave n-offset: 0,32
    const int nk  = K / BK;

    // issue async copies for k-stage kt into buffer buf
    auto issue = [&](int kt, int buf) {
        char* sb = smem + (size_t)buf * STAGE_BYTES;
        // A tile: BM x BK bf16 = 8KB -> 512 chunks of 16B, 2 per thread
#pragma unroll
        for (int i = 0; i < 2; ++i) {
            int c = t + i * 256;                   // 0..511
            int row = c >> 2;
            int off = (c & 3) * 16;
            unsigned lds = lds_addr_of(sb + (unsigned)row * 64u + off);
            uint64_t g = (uint64_t)(const char*)(Abf + (size_t)(bm + row) * K + kt * BK) + off;
            asm volatile("global_load_async_to_lds_b128 %0, %1, off"
                         :: "v"(lds), "v"(g) : "memory");
        }
        // B tile: BN x BK bf16 = 4KB -> 256 chunks of 16B, 1 per thread
        {
            int row = t >> 2;
            int off = (t & 3) * 16;
            unsigned lds = lds_addr_of(sb + A_TILE_BYTES + (unsigned)row * 64u + off);
            uint64_t g = (uint64_t)(const char*)(Bp + (size_t)(bn + row) * K + kt * BK) + off;
            asm volatile("global_load_async_to_lds_b128 %0, %1, off"
                         :: "v"(lds), "v"(g) : "memory");
        }
    };

    v8f acc00 = {}, acc01 = {}, acc10 = {}, acc11 = {};

    issue(0, 0);
    for (int kt = 0; kt < nk; ++kt) {
        asm volatile("s_wait_asynccnt 0x0" ::: "memory");
        __syncthreads();
        if (kt + 1 < nk) issue(kt + 1, (kt + 1) & 1);

        const char* sb = smem + (size_t)(kt & 1) * STAGE_BYTES;
        v16bf a0 = frag_lds(sb,                wm,      lane);
        v16bf a1 = frag_lds(sb,                wm + 16, lane);
        v16bf b0 = frag_lds(sb + A_TILE_BYTES, wn,      lane);
        v16bf b1 = frag_lds(sb + A_TILE_BYTES, wn + 16, lane);
        acc00 = __builtin_amdgcn_wmma_f32_16x16x32_bf16(false, a0, false, b0, (short)0, acc00, false, false);
        acc01 = __builtin_amdgcn_wmma_f32_16x16x32_bf16(false, a0, false, b1, (short)0, acc01, false, false);
        acc10 = __builtin_amdgcn_wmma_f32_16x16x32_bf16(false, a1, false, b0, (short)0, acc10, false, false);
        acc11 = __builtin_amdgcn_wmma_f32_16x16x32_bf16(false, a1, false, b1, (short)0, acc11, false, false);
        __syncthreads();   // tile consumed; safe to refill this buffer at kt+2
    }

    // store: element r of lane -> row = tile_m + r + 8*(lane>>4), col = tile_n + (lane&15)
    const int col0 = lane & 15;
    const int radd = (lane >> 4) << 3;
#pragma unroll
    for (int q = 0; q < 4; ++q) {
        const v8f* accp = (q == 0) ? &acc00 : (q == 1) ? &acc01 : (q == 2) ? &acc10 : &acc11;
        int mtile = (q >> 1) * 16, ntile = (q & 1) * 16;
        int col = bn + wn + ntile + col0;
        float bv = bias[col];
#pragma unroll
        for (int r = 0; r < 8; ++r) {
            int row = bm + wm + mtile + r + radd;
            float v = (*accp)[r] + bv;
            if (act == 1) v = gelu_exact(v);
            C[(size_t)row * N + col] = v;
        }
    }
}

// ---------------- embedding: circular conv3 + sinusoidal PE + time linear ----------------
__global__ void embed_kernel(const float* __restrict__ x, const float* __restrict__ xm,
                             const float* __restrict__ conv, const float* __restrict__ tw,
                             const float* __restrict__ tb, float* __restrict__ out, int L)
{
    int token = blockIdx.x;
    int b = token / L, l = token % L;
    int t = threadIdx.x;                       // 256 threads
    int lm = (l == 0) ? L - 1 : l - 1;
    int lp = (l == L - 1) ? 0 : l + 1;
    const float* x0 = x + ((size_t)b * L + lm) * 7;
    const float* x1 = x + ((size_t)b * L + l ) * 7;
    const float* x2 = x + ((size_t)b * L + lp) * 7;
    const float* mk = xm + ((size_t)b * L + l) * 4;
#pragma unroll
    for (int j = 0; j < 2; ++j) {
        int d = t + j * 256;
        float acc = tb[d];
#pragma unroll
        for (int c = 0; c < 7; ++c) {
            acc += x0[c] * conv[(0 * 7 + c) * D_MODEL + d];
            acc += x1[c] * conv[(1 * 7 + c) * D_MODEL + d];
            acc += x2[c] * conv[(2 * 7 + c) * D_MODEL + d];
        }
#pragma unroll
        for (int m = 0; m < 4; ++m) acc += mk[m] * tw[m * D_MODEL + d];
        int i = d >> 1;
        float freq = __expf(-(float)(2 * i) * (9.210340371976184f / (float)D_MODEL));
        float ang = (float)l * freq;
        acc += (d & 1) ? __cosf(ang) : __sinf(ang);
        out[(size_t)token * D_MODEL + d] = acc;
    }
}

// ---------------- layernorm (optionally fused residual) ----------------
__global__ void layernorm_kernel(const float* __restrict__ in, const float* __restrict__ resid,
                                 const float* __restrict__ g, const float* __restrict__ bb,
                                 float* __restrict__ out)
{
    __shared__ float r1[256], r2[256];
    int row = blockIdx.x, t = threadIdx.x;
    size_t off = (size_t)row * D_MODEL;
    float x0 = in[off + t]       + (resid ? resid[off + t]       : 0.f);
    float x1 = in[off + t + 256] + (resid ? resid[off + t + 256] : 0.f);
    r1[t] = x0 + x1;
    r2[t] = x0 * x0 + x1 * x1;
    __syncthreads();
    for (int s = 128; s > 0; s >>= 1) {
        if (t < s) { r1[t] += r1[t + s]; r2[t] += r2[t + s]; }
        __syncthreads();
    }
    float mean = r1[0] * (1.0f / D_MODEL);
    float var  = r2[0] * (1.0f / D_MODEL) - mean * mean;
    float rstd = rsqrtf(var + 1e-5f);
    out[off + t]       = (x0 - mean) * rstd * g[t]       + bb[t];
    out[off + t + 256] = (x1 - mean) * rstd * g[t + 256] + bb[t + 256];
}

// ---------------- sparsity measure M = max(s) - mean(s) over sampled keys ----------------
__global__ void meas_kernel(const float* __restrict__ Q, const float* __restrict__ K,
                            float* __restrict__ M, unsigned seed)
{
    int wid = blockIdx.x * 8 + (threadIdx.x >> 5);   // one wave per query
    int lane = threadIdx.x & 31;
    int l = wid % LSEQ;
    int h = (wid / LSEQ) % NH;
    int b = wid / (LSEQ * NH);
    const float* q = Q + ((size_t)(b * LSEQ + l)) * D_MODEL + h * DH;
    float mx = -1e30f, sm = 0.f;
#pragma unroll
    for (int rep = 0; rep < 2; ++rep) {
        int s = lane + rep * 32;
        if (s < UTOP) {
            unsigned kidx = (1103515245u * (unsigned)(l * 131 + s * 7919)
                             + 12345u + seed * 2654435761u) & (LSEQ - 1);
            const float* kp = K + ((size_t)(b * LSEQ + (int)kidx)) * D_MODEL + h * DH;
            float dot = 0.f;
            for (int d = 0; d < DH; ++d) dot += q[d] * kp[d];
            mx = fmaxf(mx, dot);
            sm += dot;
        }
    }
    for (int m = 16; m > 0; m >>= 1) {
        mx = fmaxf(mx, __shfl_xor(mx, m, 32));
        sm += __shfl_xor(sm, m, 32);
    }
    if (lane == 0) M[(size_t)(b * NH + h) * LSEQ + l] = mx - sm * (1.0f / LSEQ);
}

// ---------------- iterative top-UTOP per (b,h) ----------------
__global__ void topk_kernel(const float* __restrict__ M, int* __restrict__ top)
{
    __shared__ float m[LSEQ];
    __shared__ float bv[256];
    __shared__ int   bi[256];
    int bh = blockIdx.x, t = threadIdx.x;
    for (int i = t; i < LSEQ; i += 256) m[i] = M[(size_t)bh * LSEQ + i];
    __syncthreads();
    for (int it = 0; it < UTOP; ++it) {
        float best = -1e30f; int bidx = 0;
        for (int i = t; i < LSEQ; i += 256)
            if (m[i] > best) { best = m[i]; bidx = i; }
        bv[t] = best; bi[t] = bidx;
        __syncthreads();
        for (int s = 128; s > 0; s >>= 1) {
            if (t < s && bv[t + s] > bv[t]) { bv[t] = bv[t + s]; bi[t] = bi[t + s]; }
            __syncthreads();
        }
        if (t == 0) { top[bh * UTOP + it] = bi[0]; m[bi[0]] = -1e30f; }
        __syncthreads();
    }
}

// ---------------- context init: broadcast mean of V (non-causal) ----------------
__global__ void ctx_mean_kernel(const float* __restrict__ V, float* __restrict__ AT)
{
    __shared__ float part[4][DH];
    __shared__ float vm[DH];
    int bh = blockIdx.x, t = threadIdx.x;
    int b = bh / NH, h = bh % NH;
    int d = t & 63, grp = t >> 6;
    float acc = 0.f;
    for (int l = grp * 256; l < (grp + 1) * 256; ++l)
        acc += V[((size_t)(b * LSEQ + l)) * D_MODEL + h * DH + d];
    part[grp][d] = acc;
    __syncthreads();
    if (grp == 0) vm[d] = (part[0][d] + part[1][d] + part[2][d] + part[3][d]) * (1.0f / LSEQ);
    __syncthreads();
    for (int l = grp; l < LSEQ; l += 4)
        AT[((size_t)(b * LSEQ + l)) * D_MODEL + h * DH + d] = vm[d];
}

// ---------------- context init: cumsum of V along keys (causal) ----------------
__global__ void ctx_cumsum_kernel(const float* __restrict__ V, float* __restrict__ AT)
{
    int idx = blockIdx.x * blockDim.x + threadIdx.x;  // NB*NH*DH = 4096
    if (idx >= NB * NH * DH) return;
    int d = idx % DH, h = (idx / DH) % NH, b = idx / (DH * NH);
    float run = 0.f;
    for (int l = 0; l < LSEQ; ++l) {
        size_t o = ((size_t)(b * LSEQ + l)) * D_MODEL + h * DH + d;
        run += V[o];
        AT[o] = run;
    }
}

// ---------------- full attention for the selected top queries ----------------
__global__ void attn_topq_kernel(const float* __restrict__ Q, const float* __restrict__ K,
                                 const float* __restrict__ V, const int* __restrict__ top,
                                 float* __restrict__ AT, int masked)
{
    __shared__ float q[DH];
    __shared__ float sc[LSEQ];
    __shared__ float red[256];
    int bh = blockIdx.x, u = blockIdx.y, t = threadIdx.x;
    int b = bh / NH, h = bh % NH;
    int qidx = top[bh * UTOP + u];
    size_t qoff = ((size_t)(b * LSEQ + qidx)) * D_MODEL + h * DH;
    if (t < DH) q[t] = Q[qoff + t];
    __syncthreads();

    float lmax = -1e30f;
    for (int k = t; k < LSEQ; k += 256) {
        size_t ko = ((size_t)(b * LSEQ + k)) * D_MODEL + h * DH;
        float s = 0.f;
        for (int d = 0; d < DH; ++d) s += q[d] * K[ko + d];
        s *= 0.125f;                              // 1/sqrt(64)
        if (masked && k > qidx) s = -1e30f;
        sc[k] = s;
        lmax = fmaxf(lmax, s);
    }
    red[t] = lmax; __syncthreads();
    for (int s = 128; s > 0; s >>= 1) { if (t < s) red[t] = fmaxf(red[t], red[t + s]); __syncthreads(); }
    float mx = red[0];
    __syncthreads();

    float lsum = 0.f;
    for (int k = t; k < LSEQ; k += 256) {
        float e = __expf(sc[k] - mx);
        sc[k] = e;
        lsum += e;
    }
    red[t] = lsum; __syncthreads();
    for (int s = 128; s > 0; s >>= 1) { if (t < s) red[t] += red[t + s]; __syncthreads(); }
    float inv = 1.0f / red[0];
    __syncthreads();

    if (t < DH) {
        float acc = 0.f;
        for (int k = 0; k < LSEQ; ++k)
            acc += sc[k] * V[((size_t)(b * LSEQ + k)) * D_MODEL + h * DH + t];
        AT[qoff + t] = acc * inv;
    }
}

// ---------------- final projection: rows [512..1023] of each batch, 512 -> 7 ----------------
__global__ void proj_kernel(const float* __restrict__ in, const float* __restrict__ w,
                            const float* __restrict__ b, float* __restrict__ out)
{
    int idx = blockIdx.x * blockDim.x + threadIdx.x;  // 8*512*7
    if (idx >= NB * 512 * 7) return;
    int c = idx % 7, r = (idx / 7) % 512, bb = idx / (7 * 512);
    size_t row = (size_t)bb * LSEQ + 512 + r;
    float acc = b[c];
    for (int k = 0; k < D_MODEL; ++k) acc += in[row * D_MODEL + k] * w[k * 7 + c];
    out[idx] = acc;
}

// ================= host side =================
struct AttnW {
    const float *qw, *qb, *kw, *kb, *vw, *vb, *ow, *ob;
};

static void run_gemm(hipStream_t s, const float* A, const float* W, const float* bias,
                     float* C, int M, int N, int K, int act, __bf16* wpack, __bf16* apack)
{
    int totW = N * K;
    pack_w_kernel<<<(totW + 255) / 256, 256, 0, s>>>(W, wpack, K, N);
    int totA = M * K;
    pack_a_kernel<<<(totA + 255) / 256, 256, 0, s>>>(A, apack, totA);
    dim3 grid(N / BN, M / BM);
    gemm_wmma_kernel<<<grid, 256, 0, s>>>(apack, wpack, bias, C, M, N, K, act);
}

static void run_attention(hipStream_t s, const AttnW& w, const float* Xq, const float* Xkv,
                          float* Qb, float* Kb, float* Vb, float* AT, float* O,
                          float* Mbuf, int* Top, __bf16* wpack, __bf16* apack,
                          int masked, unsigned seed)
{
    run_gemm(s, Xq,  w.qw, w.qb, Qb, TOK, D_MODEL, D_MODEL, 0, wpack, apack);
    run_gemm(s, Xkv, w.kw, w.kb, Kb, TOK, D_MODEL, D_MODEL, 0, wpack, apack);
    run_gemm(s, Xkv, w.vw, w.vb, Vb, TOK, D_MODEL, D_MODEL, 0, wpack, apack);
    meas_kernel<<<(NB * NH * LSEQ) / 8, 256, 0, s>>>(Qb, Kb, Mbuf, seed);
    topk_kernel<<<NB * NH, 256, 0, s>>>(Mbuf, Top);
    if (masked)
        ctx_cumsum_kernel<<<(NB * NH * DH + 255) / 256, 256, 0, s>>>(Vb, AT);
    else
        ctx_mean_kernel<<<NB * NH, 256, 0, s>>>(Vb, AT);
    attn_topq_kernel<<<dim3(NB * NH, UTOP), 256, 0, s>>>(Qb, Kb, Vb, Top, AT, masked);
    run_gemm(s, AT, w.ow, w.ob, O, TOK, D_MODEL, D_MODEL, 0, wpack, apack);
}

extern "C" void kernel_launch(void* const* d_in, const int* in_sizes, int n_in,
                              void* d_out, int out_size, void* d_ws, size_t ws_size,
                              hipStream_t stream)
{
    (void)in_sizes; (void)n_in; (void)out_size; (void)ws_size;
    const float* x_enc      = (const float*)d_in[0];
    const float* x_mark_enc = (const float*)d_in[1];
    const float* x_dec      = (const float*)d_in[2];
    const float* x_mark_dec = (const float*)d_in[3];

    // params pytree leaves, sorted-key order
    const float* dec_emb_conv   = (const float*)d_in[4];
    const float* dec_emb_time_b = (const float*)d_in[5];
    const float* dec_emb_time_w = (const float*)d_in[6];
    AttnW dcross = { (const float*)d_in[12], (const float*)d_in[11],
                     (const float*)d_in[8],  (const float*)d_in[7],
                     (const float*)d_in[14], (const float*)d_in[13],
                     (const float*)d_in[10], (const float*)d_in[9] };
    const float* d_fc1_b = (const float*)d_in[15];
    const float* d_fc1_w = (const float*)d_in[16];
    const float* d_fc2_b = (const float*)d_in[17];
    const float* d_fc2_w = (const float*)d_in[18];
    const float* d_ln1_b = (const float*)d_in[19];
    const float* d_ln1_g = (const float*)d_in[20];
    const float* d_ln2_b = (const float*)d_in[21];
    const float* d_ln2_g = (const float*)d_in[22];
    const float* d_ln3_b = (const float*)d_in[23];
    const float* d_ln3_g = (const float*)d_in[24];
    AttnW dself = { (const float*)d_in[30], (const float*)d_in[29],
                    (const float*)d_in[26], (const float*)d_in[25],
                    (const float*)d_in[32], (const float*)d_in[31],
                    (const float*)d_in[28], (const float*)d_in[27] };
    const float* dec_norm_b = (const float*)d_in[33];
    const float* dec_norm_g = (const float*)d_in[34];
    const float* enc_emb_conv   = (const float*)d_in[35];
    const float* enc_emb_time_b = (const float*)d_in[36];
    const float* enc_emb_time_w = (const float*)d_in[37];
    const float* enc_norm_b = (const float*)d_in[70];
    const float* enc_norm_g = (const float*)d_in[71];
    const float* proj_b = (const float*)d_in[72];
    const float* proj_w = (const float*)d_in[73];

    // workspace layout
    const size_t TOKD = (size_t)TOK * D_MODEL * sizeof(float);  // 16 MB
    char* ws = (char*)d_ws;
    float* X    = (float*)(ws + 0  * TOKD);
    float* X2   = (float*)(ws + 1  * TOKD);
    float* Qb   = (float*)(ws + 2  * TOKD);
    float* Kb   = (float*)(ws + 3  * TOKD);
    float* Vb   = (float*)(ws + 4  * TOKD);
    float* AT   = (float*)(ws + 5  * TOKD);
    float* Y    = (float*)(ws + 6  * TOKD);
    float* ENC  = (float*)(ws + 7  * TOKD);
    float* MID  = (float*)(ws + 8  * TOKD);                      // 64 MB (TOK x D_FF)
    float* Mbuf = (float*)(ws + 12 * TOKD);
    int*   Top  = (int*)  (ws + 12 * TOKD + (size_t)NB * NH * LSEQ * sizeof(float));
    __bf16* WP  = (__bf16*)(ws + 12 * TOKD + (size_t)(1  << 20));   // 2 MB max (N*K bf16)
    __bf16* AP  = (__bf16*)(ws + 12 * TOKD + (size_t)(4  << 20));   // 32 MB max (TOK*D_FF bf16)

    // ---------- encoder ----------
    embed_kernel<<<TOK, 256, 0, stream>>>(x_enc, x_mark_enc, enc_emb_conv,
                                          enc_emb_time_w, enc_emb_time_b, X, LSEQ);
    for (int i = 0; i < 2; ++i) {
        int base = 38 + i * 16;
        AttnW aw = { (const float*)d_in[base + 5], (const float*)d_in[base + 4],
                     (const float*)d_in[base + 1], (const float*)d_in[base + 0],
                     (const float*)d_in[base + 7], (const float*)d_in[base + 6],
                     (const float*)d_in[base + 3], (const float*)d_in[base + 2] };
        const float* fc1_b = (const float*)d_in[base + 8];
        const float* fc1_w = (const float*)d_in[base + 9];
        const float* fc2_b = (const float*)d_in[base + 10];
        const float* fc2_w = (const float*)d_in[base + 11];
        const float* ln1_b = (const float*)d_in[base + 12];
        const float* ln1_g = (const float*)d_in[base + 13];
        const float* ln2_b = (const float*)d_in[base + 14];
        const float* ln2_g = (const float*)d_in[base + 15];

        run_attention(stream, aw, X, X, Qb, Kb, Vb, AT, Y, Mbuf, Top, WP, AP, 0, 1u + (unsigned)i);
        layernorm_kernel<<<TOK, 256, 0, stream>>>(Y, X, ln1_g, ln1_b, X2);        // h = LN1(x + attn)
        run_gemm(stream, X2, fc1_w, fc1_b, MID, TOK, D_FF, D_MODEL, 1, WP, AP);   // gelu fused
        run_gemm(stream, MID, fc2_w, fc2_b, Y, TOK, D_MODEL, D_FF, 0, WP, AP);
        layernorm_kernel<<<TOK, 256, 0, stream>>>(Y, X2, ln2_g, ln2_b, X);        // x = LN2(h + ffn)
    }
    layernorm_kernel<<<TOK, 256, 0, stream>>>(X, nullptr, enc_norm_g, enc_norm_b, ENC);

    // ---------- decoder ----------
    embed_kernel<<<TOK, 256, 0, stream>>>(x_dec, x_mark_dec, dec_emb_conv,
                                          dec_emb_time_w, dec_emb_time_b, X, LSEQ);
    // self attention (causal)
    run_attention(stream, dself, X, X, Qb, Kb, Vb, AT, Y, Mbuf, Top, WP, AP, 1, 100u);
    layernorm_kernel<<<TOK, 256, 0, stream>>>(Y, X, d_ln1_g, d_ln1_b, X2);        // x = LN1(x + self)
    // cross attention
    run_attention(stream, dcross, X2, ENC, Qb, Kb, Vb, AT, Y, Mbuf, Top, WP, AP, 0, 101u);
    layernorm_kernel<<<TOK, 256, 0, stream>>>(Y, X2, d_ln2_g, d_ln2_b, X);        // x = LN2(x + cross)
    // FFN
    run_gemm(stream, X, d_fc1_w, d_fc1_b, MID, TOK, D_FF, D_MODEL, 1, WP, AP);
    run_gemm(stream, MID, d_fc2_w, d_fc2_b, Y, TOK, D_MODEL, D_FF, 0, WP, AP);
    layernorm_kernel<<<TOK, 256, 0, stream>>>(Y, X, d_ln3_g, d_ln3_b, X2);        // x = LN3(x + ffn)
    // final norm + projection (last 512 tokens per batch)
    layernorm_kernel<<<TOK, 256, 0, stream>>>(X2, nullptr, dec_norm_g, dec_norm_b, X);
    proj_kernel<<<(NB * 512 * 7 + 255) / 256, 256, 0, stream>>>(X, proj_w, proj_b, (float*)d_out);
}
